// SDARAttention_506806141427
// MI455X (gfx1250) — matrix operand order repaired and checked by reading, not compiled, not run on hardware
//
#include <hip/hip_runtime.h>
#include <hip/hip_bf16.h>
#include <stdint.h>

// ---------------------------------------------------------------------------
// CDNA5 (gfx1250) wave32 WMMA implementation of SDAR attention layer.
//   qkv = H @ Wqkv ; RMSNorm+RoPE(q,k) ; GQA flash attention (block-causal,
//   window 1024) ; out = attn @ Wo
// Matmuls via v_wmma_f32_16x16x32_bf16 (f32 accumulate). Attention K-tiles
// move via the Tensor Data Mover (tensor_load_to_lds) with LDS double
// buffering so DMA + V-prefetch overlap the WMMA/softmax of the live chunk.
// ---------------------------------------------------------------------------

typedef __bf16 bf16_t;
typedef __attribute__((ext_vector_type(16))) bf16_t bf16x16;
typedef __attribute__((ext_vector_type(8)))  bf16_t bf16x8;
typedef __attribute__((ext_vector_type(4)))  bf16_t bf16x4;
typedef __attribute__((ext_vector_type(8)))  float  f32x8;
typedef __attribute__((ext_vector_type(4)))  unsigned int u32x4;
typedef __attribute__((ext_vector_type(8)))  int i32x8;
typedef __attribute__((ext_vector_type(4)))  int i32x4;

constexpr int Tn   = 2048;
constexpr int HIDn = 4096;
constexpr int NQn  = 32;
constexpr int NKVn = 8;
constexpr int Dn   = 128;
constexpr int WINn = 1024;
constexpr int QKVW = (NQn + 2 * NKVn) * Dn; // 6144

#ifndef __has_builtin
#define __has_builtin(x) 0
#endif
#if __has_builtin(__builtin_amdgcn_tensor_load_to_lds) && \
    __has_builtin(__builtin_amdgcn_s_wait_tensorcnt)
#define USE_TDM 1
#else
#define USE_TDM 0
#endif

#define WMMA_BF16(a, b, c) \
  __builtin_amdgcn_wmma_f32_16x16x32_bf16(false, (a), false, (b), (short)0, (c), false, false)

static __device__ __forceinline__ bf16x16 cat8(bf16x8 lo, bf16x8 hi) {
  return __builtin_shufflevector(lo, hi, 0,1,2,3,4,5,6,7,8,9,10,11,12,13,14,15);
}
static __device__ __forceinline__ bf16x8 ld8(const bf16_t* p) { return *(const bf16x8*)p; }

static __device__ __forceinline__ float halfmax(float v) { // reduce over 16-lane half
  v = fmaxf(v, __shfl_xor(v, 1, 32));
  v = fmaxf(v, __shfl_xor(v, 2, 32));
  v = fmaxf(v, __shfl_xor(v, 4, 32));
  v = fmaxf(v, __shfl_xor(v, 8, 32));
  return v;
}
static __device__ __forceinline__ float halfsum(float v) {
  v += __shfl_xor(v, 1, 32);
  v += __shfl_xor(v, 2, 32);
  v += __shfl_xor(v, 4, 32);
  v += __shfl_xor(v, 8, 32);
  return v;
}

// ---------------------------------------------------------------------------
// Generic f32-in / f32-out GEMM through bf16 WMMA.
// Block tile 128x128, K-step 64 (32 KB LDS), 256 threads = 8 wave32 (4x2),
// each wave owns a 32x64 sub-tile = 2x4 accumulators, 16 WMMA per K-stage.
// A staging packs 4 bf16 per ds_store_b64 (2x v_cvt_pk_bf16_f32); B is
// staged transposed so fragment reads are contiguous ds_load_b128 pairs.
// ---------------------------------------------------------------------------
__global__ __launch_bounds__(256) void wmma_gemm_f32bf16(
    const float* __restrict__ A, const float* __restrict__ B, float* __restrict__ C,
    int M, int N, int K)
{
  __shared__ bf16_t As [128][64];
  __shared__ bf16_t BsT[128][64];
  const int tid  = threadIdx.x;
  const int m0   = blockIdx.y * 128, n0 = blockIdx.x * 128;
  const int wave = tid >> 5, lane = tid & 31;
  const int wm   = wave & 3, wn = wave >> 2;      // 4x2 wave grid
  const int r    = lane & 15, half = lane >> 4;

  f32x8 acc[2][4] = {};

  for (int k0 = 0; k0 < K; k0 += 64) {
    __syncthreads();
#pragma unroll
    for (int i = 0; i < 8; ++i) {            // A tile: 128x64 f32 -> bf16 (packed)
      int lin = tid + 256 * i;               // 2048 float4 units
      int row = lin >> 4, c4 = (lin & 15) * 4;
      const float4 v = *(const float4*)(A + (size_t)(m0 + row) * K + k0 + c4);
      const bf16x4 pk = { (bf16_t)v.x, (bf16_t)v.y, (bf16_t)v.z, (bf16_t)v.w };
      *(bf16x4*)&As[row][c4] = pk;           // one ds_store_b64
    }
#pragma unroll
    for (int i = 0; i < 8; ++i) {            // B tile: 64x128 f32 -> bf16^T
      int lin = tid + 256 * i;
      int kr = lin >> 5, c4 = (lin & 31) * 4;
      const float4 v = *(const float4*)(B + (size_t)(k0 + kr) * N + n0 + c4);
      BsT[c4 + 0][kr] = (bf16_t)v.x; BsT[c4 + 1][kr] = (bf16_t)v.y;
      BsT[c4 + 2][kr] = (bf16_t)v.z; BsT[c4 + 3][kr] = (bf16_t)v.w;
    }
    __syncthreads();

#pragma unroll
    for (int kk = 0; kk < 64; kk += 32) {
      bf16x16 af[2];
#pragma unroll
      for (int mt = 0; mt < 2; ++mt) {       // A frag: row = lane&15, K split
        const bf16_t* p = &As[wm * 32 + mt * 16 + r][kk];
        af[mt] = cat8(ld8(p + half * 8), ld8(p + 16 + half * 8));
      }
#pragma unroll
      for (int nt = 0; nt < 4; ++nt) {       // B frag: col = lane&15, K split
        const bf16_t* p = &BsT[wn * 64 + nt * 16 + r][kk];
        bf16x16 bfr = cat8(ld8(p + half * 8), ld8(p + 16 + half * 8));
#pragma unroll
        for (int mt = 0; mt < 2; ++mt)
          acc[mt][nt] = WMMA_BF16(af[mt], bfr, acc[mt][nt]);
      }
    }
  }

#pragma unroll
  for (int mt = 0; mt < 2; ++mt)
#pragma unroll
    for (int nt = 0; nt < 4; ++nt)
#pragma unroll
      for (int i = 0; i < 8; ++i) {          // C/D: row = i + 8*half, col = r
        const int grow = m0 + wm * 32 + mt * 16 + i + 8 * half;
        const int gcol = n0 + wn * 64 + nt * 16 + r;
        C[(size_t)grow * N + gcol] = acc[mt][nt][i];
      }
}

// ---------------------------------------------------------------------------
// Fused RMSNorm + RoPE, writes bf16 Q/K/V. One 128-thread block per (t, head).
// ---------------------------------------------------------------------------
__global__ __launch_bounds__(128) void normrope_kernel(
    const float* __restrict__ qkv, const float* __restrict__ cosT, const float* __restrict__ sinT,
    const float* __restrict__ qw, const float* __restrict__ kw,
    bf16_t* __restrict__ Qo, bf16_t* __restrict__ Ko, bf16_t* __restrict__ Vo)
{
  __shared__ float partial[4];
  __shared__ float xs[128];
  const int bh = blockIdx.x % (NQn + 2 * NKVn);
  const int t  = blockIdx.x / (NQn + 2 * NKVn);
  const int d  = threadIdx.x;

  if (bh >= NQn + NKVn) {                    // V head: straight bf16 pack
    const int vh = bh - (NQn + NKVn);
    const float x = qkv[(size_t)t * QKVW + (size_t)(NQn + NKVn) * Dn + vh * Dn + d];
    Vo[(size_t)t * (NKVn * Dn) + vh * Dn + d] = (bf16_t)x;
    return;
  }
  const bool isq = bh < NQn;
  const int  h   = isq ? bh : bh - NQn;
  const size_t base = (size_t)t * QKVW + (isq ? (size_t)h * Dn : (size_t)NQn * Dn + (size_t)h * Dn);
  const float x = qkv[base + d];

  float ss = x * x;                          // mean-of-squares over D=128
  ss += __shfl_xor(ss, 1, 32);  ss += __shfl_xor(ss, 2, 32);
  ss += __shfl_xor(ss, 4, 32);  ss += __shfl_xor(ss, 8, 32);
  ss += __shfl_xor(ss, 16, 32);
  if ((threadIdx.x & 31) == 0) partial[threadIdx.x >> 5] = ss;
  __syncthreads();
  const float tot = partial[0] + partial[1] + partial[2] + partial[3];
  const float inv = rsqrtf(tot * (1.0f / 128.0f) + 1e-6f);
  const float xn  = x * inv * (isq ? qw[d] : kw[d]);
  xs[d] = xn;
  __syncthreads();
  const float rot = (d < 64) ? -xs[d + 64] : xs[d - 64];
  const float o   = xn * cosT[(size_t)t * Dn + d] + rot * sinT[(size_t)t * Dn + d];
  if (isq) Qo[(size_t)t * (NQn * Dn) + h * Dn + d] = (bf16_t)o;
  else     Ko[(size_t)t * (NKVn * Dn) + h * Dn + d] = (bf16_t)o;
}

// ---------------------------------------------------------------------------
// Flash attention. Block = (q-tile of 64 rows, head h), 128 threads = 4 waves,
// wave w owns q rows [tile*64 + w*16, +16). Keys processed in chunks of 32
// with LDS double buffering:
//   - Kt[buf][32][128] filled by TDM (tensor_load_to_lds), issued one chunk
//     ahead so the DMA overlaps compute (wave 0 issues; TENSORcnt per-wave).
//   - VtT[buf][128][32] (transposed for PV B-fragments): next chunk's V is
//     global-loaded into registers before compute, stored to LDS after.
// QK^T: A = Q 16x32(d), B = K^T 32(d)x16(key)   -> 8 wmma / chunk
// PV:   A = P 16x32(key), B = V 32(key)x16(d)   -> 8 wmma / chunk
// ---------------------------------------------------------------------------
__global__ __launch_bounds__(128) void attn_wmma(
    const bf16_t* __restrict__ Q, const bf16_t* __restrict__ Kc, const bf16_t* __restrict__ Vc,
    float* __restrict__ O)
{
  __shared__ bf16_t Kt [2][32][128];
  __shared__ bf16_t VtT[2][128][32];
  __shared__ bf16_t Pbuf[4][16][32];
  const int tid  = threadIdx.x;
  const int wave = tid >> 5, lane = tid & 31, r = lane & 15, half = lane >> 4;
  const int h    = blockIdx.y, kvh = h >> 2;  // GQA: 4 q-heads per kv-head
  const int tile = blockIdx.x;
  const int q0   = tile * 64 + wave * 16;
  const int key  = tid >> 2, seg0 = (tid & 3) * 32;  // V-stage: 128 thr x 1 row-quarter

#if USE_TDM
  const unsigned ldsKt0 =
      (unsigned)(uintptr_t)(__attribute__((address_space(3))) const void*)&Kt[0][0][0];
  const unsigned ldsKt1 =
      (unsigned)(uintptr_t)(__attribute__((address_space(3))) const void*)&Kt[1][0][0];
#endif

  bf16x16 qf[4];                              // Q 16x128 as 4 A-fragments
  {
    const bf16_t* qp = Q + (size_t)(q0 + r) * (NQn * Dn) + h * Dn;
#pragma unroll
    for (int dc = 0; dc < 4; ++dc) {
      const bf16_t* p = qp + dc * 32;
      qf[dc] = cat8(ld8(p + half * 8), ld8(p + 16 + half * 8));
    }
  }

  float m_run[8], l_run[8];
  f32x8 oacc[8] = {};
#pragma unroll
  for (int i = 0; i < 8; ++i) { m_run[i] = -1e30f; l_run[i] = 0.0f; }

  int kmin = tile * 64 - (WINn - 1); if (kmin < 0) kmin = 0;
  const int kstart = kmin & ~31;              // chunk-aligned; edges masked
  const int kend   = tile * 64 + 64;          // block-causal upper bound
  const float scale = 0.088388347648318447f;  // D^-0.5

  // ---- prologue: stage chunk 0 into buffer 0 --------------------------------
  {
#if USE_TDM
    if (wave == 0) {
      const uint64_t ga =
          (uint64_t)(uintptr_t)(Kc + (size_t)kstart * (NKVn * Dn) + (size_t)kvh * Dn);
      const u32x4 g0 = { 1u, ldsKt0, (unsigned)ga, (unsigned)(ga >> 32) | 0x80000000u };
      const i32x8 g1 = { (1 << 16), (Dn & 0xFFFF) << 16, (32 << 16), (Dn << 16),
                         32, NKVn * Dn, 0, 0 };
      const i32x4 g2 = { 0, 0, 0, 0 };
      const i32x4 g3 = { 0, 0, 0, 0 };
      const i32x8 g4 = { 0, 0, 0, 0, 0, 0, 0, 0 };
      __builtin_amdgcn_tensor_load_to_lds(g0, g1, g2, g3, g4, 0);
    }
#endif
#pragma unroll
    for (int it = 0; it < 4; ++it) {          // V chunk 0 (+K if no TDM)
      const int unit = tid + 128 * it;
      const int ky = unit >> 4, sg = (unit & 15) * 8;
      const size_t src = (size_t)(kstart + ky) * (NKVn * Dn) + kvh * Dn + sg;
#if !USE_TDM
      *(bf16x8*)&Kt[0][ky][sg] = ld8(Kc + src);
#endif
      bf16x8 vv = ld8(Vc + src);
#pragma unroll
      for (int j = 0; j < 8; ++j) VtT[0][sg + j][ky] = vv[j];
    }
#if USE_TDM
    if (wave == 0) __builtin_amdgcn_s_wait_tensorcnt(0);
#endif
    __syncthreads();
  }

  int buf = 0;
  for (int kc = kstart; kc < kend; kc += 32) {
    const int  nb      = buf ^ 1;
    const bool hasnext = (kc + 32) < kend;

    // ---- kick off next chunk: TDM for K, register prefetch for V ------------
    bf16x8 vreg[4];
#if !USE_TDM
    bf16x8 kreg[4];
#endif
    if (hasnext) {
#if USE_TDM
      if (wave == 0) {
        const uint64_t ga =
            (uint64_t)(uintptr_t)(Kc + (size_t)(kc + 32) * (NKVn * Dn) + (size_t)kvh * Dn);
        const u32x4 g0 = { 1u, (nb ? ldsKt1 : ldsKt0), (unsigned)ga,
                           (unsigned)(ga >> 32) | 0x80000000u };
        const i32x8 g1 = { (1 << 16), (Dn & 0xFFFF) << 16, (32 << 16), (Dn << 16),
                           32, NKVn * Dn, 0, 0 };
        const i32x4 g2 = { 0, 0, 0, 0 };
        const i32x4 g3 = { 0, 0, 0, 0 };
        const i32x8 g4 = { 0, 0, 0, 0, 0, 0, 0, 0 };
        __builtin_amdgcn_tensor_load_to_lds(g0, g1, g2, g3, g4, 0);
      }
#endif
#pragma unroll
      for (int it = 0; it < 4; ++it) {
        const int unit = tid + 128 * it;
        const int ky = unit >> 4, sg = (unit & 15) * 8;
        const size_t src = (size_t)(kc + 32 + ky) * (NKVn * Dn) + kvh * Dn + sg;
        vreg[it] = ld8(Vc + src);
#if !USE_TDM
        kreg[it] = ld8(Kc + src);
#endif
      }
    }

    // ---- compute on current buffer -----------------------------------------
    f32x8 s0 = {}, s1 = {};                   // two 16x16 score tiles
#pragma unroll
    for (int dc = 0; dc < 4; ++dc) {
      const bf16_t* p0 = &Kt[buf][r][dc * 32 + half * 8];
      const bf16_t* p1 = &Kt[buf][16 + r][dc * 32 + half * 8];
      bf16x16 b0 = cat8(ld8(p0), ld8(p0 + 16));
      bf16x16 b1 = cat8(ld8(p1), ld8(p1 + 16));
      s0 = WMMA_BF16(qf[dc], b0, s0);
      s1 = WMMA_BF16(qf[dc], b1, s1);
    }

#pragma unroll
    for (int i = 0; i < 8; ++i) {             // mask + online softmax per row
      const int qq  = q0 + i + 8 * half;
      const int k0i = kc + r, k1i = kc + 16 + r;
      const float v0 = (((k0i >> 2) <= (qq >> 2)) && (qq - k0i < WINn)) ? s0[i] * scale : -1e30f;
      const float v1 = (((k1i >> 2) <= (qq >> 2)) && (qq - k1i < WINn)) ? s1[i] * scale : -1e30f;
      const float rowm = halfmax(fmaxf(v0, v1));
      const float mnew = fmaxf(m_run[i], rowm);
      const float p0 = __expf(v0 - mnew);
      const float p1 = __expf(v1 - mnew);
      const float rs = halfsum(p0 + p1);
      const float sf = __expf(m_run[i] - mnew);
      m_run[i] = mnew;
      l_run[i] = l_run[i] * sf + rs;
#pragma unroll
      for (int dt = 0; dt < 8; ++dt) oacc[dt][i] *= sf;
      const int row = i + 8 * half;           // C-layout -> LDS for transpose
      Pbuf[wave][row][r]      = (bf16_t)p0;
      Pbuf[wave][row][16 + r] = (bf16_t)p1;
    }
    // wave-private LDS round-trip: order ds stores before ds loads (CDNA5
    // split counters; DS ops of one wave complete in order).
    asm volatile("s_wait_dscnt 0" ::: "memory");

    const bf16_t* pp = &Pbuf[wave][r][0];     // P as A-fragment (16x32 keys)
    bf16x16 pf = cat8(ld8(pp + half * 8), ld8(pp + 16 + half * 8));
#pragma unroll
    for (int dt = 0; dt < 8; ++dt) {          // PV accumulate over 8 d-tiles
      const bf16_t* vp = &VtT[buf][dt * 16 + r][half * 8];
      bf16x16 vf = cat8(ld8(vp), ld8(vp + 16));
      oacc[dt] = WMMA_BF16(pf, vf, oacc[dt]);
    }

    // ---- drain next-chunk prefetch into the back buffer ---------------------
    if (hasnext) {
#pragma unroll
      for (int it = 0; it < 4; ++it) {
        const int unit = tid + 128 * it;
        const int ky = unit >> 4, sg = (unit & 15) * 8;
#if !USE_TDM
        *(bf16x8*)&Kt[nb][ky][sg] = kreg[it];
#endif
#pragma unroll
        for (int j = 0; j < 8; ++j) VtT[nb][sg + j][ky] = vreg[it][j];
      }
#if USE_TDM
      if (wave == 0) __builtin_amdgcn_s_wait_tensorcnt(0);
#endif
    }
    __syncthreads();
    buf = nb;
  }

#pragma unroll
  for (int dt = 0; dt < 8; ++dt)
#pragma unroll
    for (int i = 0; i < 8; ++i) {
      const int t = q0 + i + 8 * half;
      O[(size_t)t * (NQn * Dn) + h * Dn + dt * 16 + r] = oacc[dt][i] / l_run[i];
    }
}

// ---------------------------------------------------------------------------
// Host-side launcher. Workspace layout (≈109 MB):
//   qkv f32 [T][6144] | Q bf16 [T][NQ][D] | K bf16 [T][NKV][D] |
//   V bf16 [T][NKV][D] | attn f32 [T][NQ*D]
// ---------------------------------------------------------------------------
extern "C" void kernel_launch(void* const* d_in, const int* in_sizes, int n_in,
                              void* d_out, int out_size, void* d_ws, size_t ws_size,
                              hipStream_t stream)
{
  (void)in_sizes; (void)n_in; (void)out_size; (void)ws_size;
  const float* hidden = (const float*)d_in[0];
  const float* cosT   = (const float*)d_in[1];
  const float* sinT   = (const float*)d_in[2];
  const float* wqkv   = (const float*)d_in[3];
  const float* qw     = (const float*)d_in[4];
  const float* kw     = (const float*)d_in[5];
  const float* wo     = (const float*)d_in[6];
  float* out = (float*)d_out;

  char* ws = (char*)d_ws;
  float*  qkv  = (float*)ws;   ws += (size_t)Tn * QKVW      * sizeof(float);
  bf16_t* Qb   = (bf16_t*)ws;  ws += (size_t)Tn * NQn * Dn  * sizeof(bf16_t);
  bf16_t* Kb   = (bf16_t*)ws;  ws += (size_t)Tn * NKVn * Dn * sizeof(bf16_t);
  bf16_t* Vb   = (bf16_t*)ws;  ws += (size_t)Tn * NKVn * Dn * sizeof(bf16_t);
  float*  attn = (float*)ws;

  wmma_gemm_f32bf16<<<dim3(QKVW / 128, Tn / 128), 256, 0, stream>>>(
      hidden, wqkv, qkv, Tn, QKVW, HIDn);
  normrope_kernel<<<Tn * (NQn + 2 * NKVn), 128, 0, stream>>>(
      qkv, cosT, sinT, qw, kw, Qb, Kb, Vb);
  attn_wmma<<<dim3(Tn / 64, NQn), 128, 0, stream>>>(Qb, Kb, Vb, attn);
  wmma_gemm_f32bf16<<<dim3(HIDn / 128, Tn / 128), 256, 0, stream>>>(
      attn, wo, out, Tn, HIDn, NQn * Dn);
}
